// MessagePassing_38199439131004
// MI455X (gfx1250) — matrix operand order repaired and checked by reading, not compiled, not run on hardware
//
#include <hip/hip_runtime.h>
#include <hip/hip_bf16.h>

typedef _Float16 v16h __attribute__((ext_vector_type(16)));
typedef _Float16 v8h  __attribute__((ext_vector_type(8)));
typedef float    v8f  __attribute__((ext_vector_type(8)));
typedef float    v4f  __attribute__((ext_vector_type(4)));

#define N_NODES 50000
#define N_EDGES 800000
#define N_GRAPHS 500

__device__ __forceinline__ float selu_f(float v) {
    const float kA = 1.6732632423543772f, kS = 1.0507009873554805f;
    return kS * (v > 0.f ? v : kA * (__expf(v) - 1.f));
}
__device__ __forceinline__ float sigmoid_f(float v) {
    return 1.f / (1.f + __expf(-v));
}
__device__ __forceinline__ void atomAddF(float* p, float v) {
    unsafeAtomicAdd(p, v);  // global_atomic_add_f32 (no return)
}

// ---------------- weight pre-swizzle into B-fragment layout -----------------
// B layout (16-bit, 32x16 per chunk): lane L holds col n=L&15, slot t -> k=(t>>3)*16+(L>>4)*8+(t&7)
// Buffer layout: idx = ((q*NTILES + tN)*32 + L)*16 + t

__global__ void k_swz_edgeB(const float* __restrict__ W_l2, const float* __restrict__ W_b2,
                            _Float16* __restrict__ dst) {
    int idx = blockIdx.x * blockDim.x + threadIdx.x;
    if (idx >= 8 * 17 * 512) return;
    int t = idx & 15, L = (idx >> 4) & 31;
    int rem = idx >> 9;
    int tN = rem % 17, q = rem / 17;
    int k = q * 32 + ((t >> 3) << 4) + ((L >> 4) << 3) + (t & 7);
    int nl = L & 15;
    float v = (tN < 16) ? W_l2[k * 256 + tN * 16 + nl] : W_b2[k * 16 + nl];
    dst[idx] = (_Float16)v;
}

__global__ void k_swz_B(const float* __restrict__ src, int Kact, int nChunks,
                        _Float16* __restrict__ dst) {
    int idx = blockIdx.x * blockDim.x + threadIdx.x;
    if (idx >= nChunks * 8 * 512) return;
    int t = idx & 15, L = (idx >> 4) & 31, tN = (idx >> 9) & 7, q = idx >> 12;
    int k = q * 32 + ((t >> 3) << 4) + ((L >> 4) << 3) + (t & 7);
    int col = tN * 16 + (L & 15);
    dst[idx] = (_Float16)((k < Kact) ? src[k * 128 + col] : 0.f);
}

// ---------------- init / zero -----------------
__global__ void k_init_h(const float* __restrict__ x, float* __restrict__ h, int n16) {
    int idx = blockIdx.x * blockDim.x + threadIdx.x;
    if (idx >= n16) return;
    int node = idx >> 4, j = idx & 15;
    h[idx] = (j < 2) ? x[node * 2 + j] : 0.f;
}
__global__ void k_zero(float* __restrict__ p, int n) {
    int idx = blockIdx.x * blockDim.x + threadIdx.x;
    if (idx < n) p[idx] = 0.f;
}

// ---------------- edge MLP: a = selu(e*Wl1+bl1)@Wl2+bl2 ; be = selu(e*Wb1+bb1)@Wb2+bb2
// one wave per 16 edges; WMMA f32_16x16x32_f16; result tile staged in LDS and
// written out with wide b128 nontemporal stores (row-major [edge][256] f16).
__global__ void __launch_bounds__(128)
k_edge_mlp(const float* __restrict__ e,
           const float* __restrict__ W_l1, const float* __restrict__ b_l1,
           const float* __restrict__ W_b1, const float* __restrict__ b_b1,
           const float* __restrict__ b_l2, const float* __restrict__ b_b2,
           const _Float16* __restrict__ WB,
           _Float16* __restrict__ aOut, float* __restrict__ beOut) {
    __shared__ float sWl1[256], sbl1[256], sWb1[256], sbb1[256], sbl2[256], sbb2[16];
    __shared__ __attribute__((aligned(16))) _Float16 sA[4 * 4096];  // 4 waves x (16x256)
    int tid = threadIdx.x;
    for (int i = tid; i < 256; i += 128) {
        sWl1[i] = W_l1[i]; sbl1[i] = b_l1[i]; sWb1[i] = W_b1[i];
        sbb1[i] = b_b1[i]; sbl2[i] = b_l2[i];
    }
    if (tid < 16) sbb2[tid] = b_b2[tid];
    __syncthreads();

    int wave = tid >> 5, lane = tid & 31;
    int ebase = (blockIdx.x * 4 + wave) * 16;
    int m = lane & 15, hi = lane >> 4, nl = m;
    float ev = e[ebase + m];

    // A fragments for l-hidden: slot t of lane -> k = q*32 + (t/8)*16 + hi*8 + (t%8)
    v16h aq[8];
#pragma unroll
    for (int q = 0; q < 8; q++) {
#pragma unroll
        for (int t = 0; t < 16; t++) {
            int k = q * 32 + ((t >> 3) << 4) + (hi << 3) + (t & 7);
            aq[q][t] = (_Float16)selu_f(ev * sWl1[k] + sbl1[k]);
        }
    }
    _Float16* myA = sA + wave * 4096;
    for (int tN = 0; tN < 16; tN++) {
        float bias = sbl2[tN * 16 + nl];
        v8f acc;
#pragma unroll
        for (int v = 0; v < 8; v++) acc[v] = bias;
#pragma unroll
        for (int q = 0; q < 8; q++) {
            const v16h bq = *(const v16h*)(WB + (size_t)(((q * 17 + tN) * 32 + lane)) * 16);
            acc = __builtin_amdgcn_wmma_f32_16x16x32_f16(false, aq[q], false, bq,
                                                         (short)0, acc, false, false);
        }
#pragma unroll
        for (int v = 0; v < 8; v++) {
            int row = v + hi * 8;
            myA[row * 256 + tN * 16 + nl] = (_Float16)acc[v];
        }
    }
    // b-hidden -> be (tile 16)
#pragma unroll
    for (int q = 0; q < 8; q++) {
#pragma unroll
        for (int t = 0; t < 16; t++) {
            int k = q * 32 + ((t >> 3) << 4) + (hi << 3) + (t & 7);
            aq[q][t] = (_Float16)selu_f(ev * sWb1[k] + sbb1[k]);
        }
    }
    {
        float bias = sbb2[nl];
        v8f acc;
#pragma unroll
        for (int v = 0; v < 8; v++) acc[v] = bias;
#pragma unroll
        for (int q = 0; q < 8; q++) {
            const v16h bq = *(const v16h*)(WB + (size_t)(((q * 17 + 16) * 32 + lane)) * 16);
            acc = __builtin_amdgcn_wmma_f32_16x16x32_f16(false, aq[q], false, bq,
                                                         (short)0, acc, false, false);
        }
#pragma unroll
        for (int v = 0; v < 8; v++) {
            int row = v + hi * 8;
            beOut[(size_t)(ebase + row) * 16 + nl] = acc[v];
        }
    }
    __syncthreads();
    // bulk copy LDS tile -> global, 16B lanes: lane L covers edge L>>1, half L&1
    {
        int erow = lane >> 1, half = lane & 1;
        const v8h* srcp = (const v8h*)(myA + erow * 256 + half * 128);
        v8h* dstp = (v8h*)(aOut + (size_t)(ebase + erow) * 256 + half * 128);
#pragma unroll
        for (int c = 0; c < 16; c++) __builtin_nontemporal_store(srcp[c], dstp + c);
    }
}

// ---------------- per-pass edge message: m[second] += a_e @ h[first] + be_e ----
__global__ void k_edge_msg(const _Float16* __restrict__ a, const float* __restrict__ be,
                           const float* __restrict__ h, const int* __restrict__ first,
                           const int* __restrict__ second, float* __restrict__ mOut, int E) {
    int eid = blockIdx.x * blockDim.x + threadIdx.x;
    if (eid >= E) return;
    int src = first[eid], dst = second[eid];
    const v4f* hp = (const v4f*)(h + (size_t)src * 16);
    v4f h0 = hp[0], h1 = hp[1], h2 = hp[2], h3 = hp[3];
    float hg[16] = {h0.x,h0.y,h0.z,h0.w, h1.x,h1.y,h1.z,h1.w,
                    h2.x,h2.y,h2.z,h2.w, h3.x,h3.y,h3.z,h3.w};
    const v8h* arp = (const v8h*)(a + (size_t)eid * 256);
    const v4f* bep = (const v4f*)(be + (size_t)eid * 16);
    v4f bv[4] = {__builtin_nontemporal_load(bep + 0), __builtin_nontemporal_load(bep + 1),
                 __builtin_nontemporal_load(bep + 2), __builtin_nontemporal_load(bep + 3)};
    const float* bvf = (const float*)bv;
#pragma unroll
    for (int i = 0; i < 16; i++) {
        v8h lo = __builtin_nontemporal_load(arp + 2 * i);
        v8h hiv = __builtin_nontemporal_load(arp + 2 * i + 1);
        float s = bvf[i];
#pragma unroll
        for (int j = 0; j < 8; j++) s += (float)lo[j] * hg[j] + (float)hiv[j] * hg[8 + j];
        atomAddF(&mOut[(size_t)dst * 16 + i], s);
    }
}

// ---------------- GRU (reset_after=True); 16 threads per node ------------------
__global__ void k_gru(const float* __restrict__ m, float* __restrict__ h,
                      const float* __restrict__ Wx, const float* __restrict__ Uh,
                      const float* __restrict__ bx, const float* __restrict__ bh, int N) {
    __shared__ float sWx[768], sUh[768], sbx[48], sbh[48], sm[256], sh[256];
    int tid = threadIdx.x;
    for (int i = tid; i < 768; i += 256) { sWx[i] = Wx[i]; sUh[i] = Uh[i]; }
    if (tid < 48) { sbx[tid] = bx[tid]; sbh[tid] = bh[tid]; }
    int node = blockIdx.x * 16 + (tid >> 4);
    int ch = tid & 15;
    bool ok = node < N;
    sm[tid] = ok ? m[(size_t)node * 16 + ch] : 0.f;
    sh[tid] = ok ? h[(size_t)node * 16 + ch] : 0.f;
    __syncthreads();
    if (!ok) return;
    int r0 = (tid >> 4) << 4;
    float xz = sbx[ch], xr = sbx[16 + ch], xh = sbx[32 + ch];
    float hz = sbh[ch], hr = sbh[16 + ch], hhb = sbh[32 + ch];
#pragma unroll
    for (int k = 0; k < 16; k++) {
        float mk = sm[r0 + k], hk = sh[r0 + k];
        xz += mk * sWx[k * 48 + ch]; xr += mk * sWx[k * 48 + 16 + ch]; xh += mk * sWx[k * 48 + 32 + ch];
        hz += hk * sUh[k * 48 + ch]; hr += hk * sUh[k * 48 + 16 + ch]; hhb += hk * sUh[k * 48 + 32 + ch];
    }
    float z = sigmoid_f(xz + hz);
    float r = sigmoid_f(xr + hr);
    float hh = tanhf(xh + r * hhb);
    float hold = sh[tid];
    h[(size_t)node * 16 + ch] = z * hold + (1.f - z) * hh;
}

// ---------------- readout: per 16-node tile, WMMA through LDS ------------------
__global__ void k_readout(const float* __restrict__ h, const float* __restrict__ x,
                          const int* __restrict__ segment,
                          const _Float16* __restrict__ Bi, const _Float16* __restrict__ Bj1,
                          const _Float16* __restrict__ BR, const _Float16* __restrict__ Bj2,
                          const float* __restrict__ b_i, const float* __restrict__ b_R,
                          const float* __restrict__ b_j1, const float* __restrict__ b_j2,
                          float* __restrict__ nb, int N) {
    __shared__ _Float16 sI[4][2048];
    __shared__ _Float16 sJ[4][2048];
    int wave = threadIdx.x >> 5, lane = threadIdx.x & 31;
    int tile = blockIdx.x * 4 + wave;
    bool active = (tile * 16) < N;
    int m = lane & 15, hi = lane >> 4, nl = m;

    // hx = [h | x | 0pad] as A-frag (K=32, one chunk)
    v16h hxA;
#pragma unroll
    for (int t = 0; t < 16; t++) {
        int k = ((t >> 3) << 4) + (hi << 3) + (t & 7);
        float v = 0.f;
        if (active) {
            int node = tile * 16 + m;
            if (k < 16) v = h[(size_t)node * 16 + k];
            else if (k < 18) v = x[(size_t)node * 2 + (k - 16)];
        }
        hxA[t] = (_Float16)v;
    }
    // layer1: i = tanh(hx@W_i+b_i), jmid = selu(hx@W_j1+b_j1) -> LDS (f16, row-major 16x128)
    for (int tN = 0; tN < 8; tN++) {
        float bi = b_i[tN * 16 + nl], bj = b_j1[tN * 16 + nl];
        v8f ai, aj;
#pragma unroll
        for (int v = 0; v < 8; v++) { ai[v] = bi; aj[v] = bj; }
        const v16h b1 = *(const v16h*)(Bi + (size_t)((tN * 32 + lane)) * 16);
        const v16h b2 = *(const v16h*)(Bj1 + (size_t)((tN * 32 + lane)) * 16);
        ai = __builtin_amdgcn_wmma_f32_16x16x32_f16(false, hxA, false, b1, (short)0, ai, false, false);
        aj = __builtin_amdgcn_wmma_f32_16x16x32_f16(false, hxA, false, b2, (short)0, aj, false, false);
#pragma unroll
        for (int v = 0; v < 8; v++) {
            int row = v + hi * 8;
            sI[wave][row * 128 + tN * 16 + nl] = (_Float16)tanhf(ai[v]);
            sJ[wave][row * 128 + tN * 16 + nl] = (_Float16)selu_f(aj[v]);
        }
    }
    __syncthreads();
    // reload as A-frags (K=128 -> 4 chunks)
    v16h iA[4], jA[4];
#pragma unroll
    for (int q = 0; q < 4; q++) {
#pragma unroll
        for (int t = 0; t < 16; t++) {
            int k = q * 32 + ((t >> 3) << 4) + (hi << 3) + (t & 7);
            iA[q][t] = sI[wave][m * 128 + k];
            jA[q][t] = sJ[wave][m * 128 + k];
        }
    }
    // layer2: RR=sigmoid(i@W_R+b_R); jf=jmid@W_j2+b_j2; nb[seg] += RR*jf
    for (int tN = 0; tN < 8; tN++) {
        float br = b_R[tN * 16 + nl], bj = b_j2[tN * 16 + nl];
        v8f aR, aJ;
#pragma unroll
        for (int v = 0; v < 8; v++) { aR[v] = br; aJ[v] = bj; }
#pragma unroll
        for (int q = 0; q < 4; q++) {
            const v16h bR = *(const v16h*)(BR + (size_t)(((q * 8 + tN) * 32 + lane)) * 16);
            const v16h bJ = *(const v16h*)(Bj2 + (size_t)(((q * 8 + tN) * 32 + lane)) * 16);
            aR = __builtin_amdgcn_wmma_f32_16x16x32_f16(false, iA[q], false, bR, (short)0, aR, false, false);
            aJ = __builtin_amdgcn_wmma_f32_16x16x32_f16(false, jA[q], false, bJ, (short)0, aJ, false, false);
        }
        if (active) {
#pragma unroll
            for (int v = 0; v < 8; v++) {
                int node = tile * 16 + v + hi * 8;
                int g = segment[node];
                float val = sigmoid_f(aR[v]) * aJ[v];
                atomAddF(&nb[(size_t)g * 128 + tN * 16 + nl], val);
            }
        }
    }
}

// ---------------- final head ------------------
__global__ void k_head1(const float* __restrict__ nb, const float* __restrict__ W_f1,
                        const float* __restrict__ b_f1, float* __restrict__ t1) {
    __shared__ float srow[128];
    int g = blockIdx.x, k = threadIdx.x;
    srow[k] = nb[(size_t)g * 128 + k];
    __syncthreads();
    float s = b_f1[k];
#pragma unroll 8
    for (int j = 0; j < 128; j++) s += srow[j] * W_f1[j * 128 + k];
    t1[(size_t)g * 128 + k] = selu_f(s);
}
__global__ void k_head2(const float* __restrict__ t1, const float* __restrict__ W_f2,
                        const float* __restrict__ b_f2, float* __restrict__ out, int G) {
    int g = blockIdx.x * blockDim.x + threadIdx.x;
    if (g >= G) return;
    float s = b_f2[0];
#pragma unroll 8
    for (int k = 0; k < 128; k++) s += t1[(size_t)g * 128 + k] * W_f2[k];
    out[g] = s;
}

extern "C" void kernel_launch(void* const* d_in, const int* in_sizes, int n_in,
                              void* d_out, int out_size, void* d_ws, size_t ws_size,
                              hipStream_t stream) {
    const float* x     = (const float*)d_in[0];
    const float* e     = (const float*)d_in[1];
    const float* W_l1  = (const float*)d_in[2];
    const float* b_l1  = (const float*)d_in[3];
    const float* W_l2  = (const float*)d_in[4];
    const float* b_l2  = (const float*)d_in[5];
    const float* W_b1  = (const float*)d_in[6];
    const float* b_b1  = (const float*)d_in[7];
    const float* W_b2  = (const float*)d_in[8];
    const float* b_b2  = (const float*)d_in[9];
    const float* gWx   = (const float*)d_in[10];
    const float* gUh   = (const float*)d_in[11];
    const float* gbx   = (const float*)d_in[12];
    const float* gbh   = (const float*)d_in[13];
    const float* W_i   = (const float*)d_in[14];
    const float* b_i   = (const float*)d_in[15];
    const float* W_R   = (const float*)d_in[16];
    const float* b_R   = (const float*)d_in[17];
    const float* W_j1  = (const float*)d_in[18];
    const float* b_j1  = (const float*)d_in[19];
    const float* W_j2  = (const float*)d_in[20];
    const float* b_j2  = (const float*)d_in[21];
    const float* W_f1  = (const float*)d_in[22];
    const float* b_f1  = (const float*)d_in[23];
    const float* W_f2  = (const float*)d_in[24];
    const float* b_f2  = (const float*)d_in[25];
    const int* first   = (const int*)d_in[26];
    const int* second  = (const int*)d_in[27];
    const int* segment = (const int*)d_in[28];
    float* out = (float*)d_out;

    char* ws = (char*)d_ws;
    size_t off = 0;
    auto take = [&](size_t bytes) { size_t o = off; off = (off + bytes + 255) & ~(size_t)255; return o; };
    _Float16* aBuf  = (_Float16*)(ws + take((size_t)N_EDGES * 256 * sizeof(_Float16)));
    float*    beBuf = (float*)   (ws + take((size_t)N_EDGES * 16 * sizeof(float)));
    float*    hBuf  = (float*)   (ws + take((size_t)N_NODES * 16 * sizeof(float)));
    float*    mBuf  = (float*)   (ws + take((size_t)N_NODES * 16 * sizeof(float)));
    _Float16* WBe   = (_Float16*)(ws + take((size_t)8 * 17 * 512 * sizeof(_Float16)));
    _Float16* Bi    = (_Float16*)(ws + take((size_t)1 * 8 * 512 * sizeof(_Float16)));
    _Float16* Bj1   = (_Float16*)(ws + take((size_t)1 * 8 * 512 * sizeof(_Float16)));
    _Float16* BR    = (_Float16*)(ws + take((size_t)4 * 8 * 512 * sizeof(_Float16)));
    _Float16* Bj2   = (_Float16*)(ws + take((size_t)4 * 8 * 512 * sizeof(_Float16)));
    float*    nbBuf = (float*)   (ws + take((size_t)N_GRAPHS * 128 * sizeof(float)));
    float*    t1Buf = (float*)   (ws + take((size_t)N_GRAPHS * 128 * sizeof(float)));

    // weight pre-swizzle
    k_swz_edgeB<<<(8 * 17 * 512 + 255) / 256, 256, 0, stream>>>(W_l2, W_b2, WBe);
    k_swz_B<<<(1 * 8 * 512 + 255) / 256, 256, 0, stream>>>(W_i, 18, 1, Bi);
    k_swz_B<<<(1 * 8 * 512 + 255) / 256, 256, 0, stream>>>(W_j1, 18, 1, Bj1);
    k_swz_B<<<(4 * 8 * 512 + 255) / 256, 256, 0, stream>>>(W_R, 128, 4, BR);
    k_swz_B<<<(4 * 8 * 512 + 255) / 256, 256, 0, stream>>>(W_j2, 128, 4, Bj2);

    // init
    k_init_h<<<(N_NODES * 16 + 255) / 256, 256, 0, stream>>>(x, hBuf, N_NODES * 16);
    k_zero<<<(N_GRAPHS * 128 + 255) / 256, 256, 0, stream>>>(nbBuf, N_GRAPHS * 128);

    // edge MLPs (hoisted, WMMA): 4 waves/block, 16 edges/wave
    k_edge_mlp<<<N_EDGES / 64, 128, 0, stream>>>(e, W_l1, b_l1, W_b1, b_b1, b_l2, b_b2,
                                                 WBe, aBuf, beBuf);

    // 8 message-passing iterations
    for (int p = 0; p < 8; p++) {
        k_zero<<<(N_NODES * 16 + 255) / 256, 256, 0, stream>>>(mBuf, N_NODES * 16);
        k_edge_msg<<<(N_EDGES + 255) / 256, 256, 0, stream>>>(aBuf, beBuf, hBuf, first,
                                                              second, mBuf, N_EDGES);
        k_gru<<<(N_NODES + 15) / 16, 256, 0, stream>>>(mBuf, hBuf, gWx, gUh, gbx, gbh, N_NODES);
    }

    // readout (WMMA) + head
    int tiles = (N_NODES + 15) / 16;
    k_readout<<<(tiles + 3) / 4, 128, 0, stream>>>(hBuf, x, segment, Bi, Bj1, BR, Bj2,
                                                   b_i, b_R, b_j1, b_j2, nbBuf, N_NODES);
    k_head1<<<N_GRAPHS, 128, 0, stream>>>(nbBuf, W_f1, b_f1, t1Buf);
    k_head2<<<(N_GRAPHS + 255) / 256, 256, 0, stream>>>(t1Buf, W_f2, b_f2, out, N_GRAPHS);
}
